// Net_53163105190518
// MI455X (gfx1250) — compile-verified, matrix operand order checked by reading, not run on hardware
//
#include <hip/hip_runtime.h>

#define DIM  16
#define BOND 4
#define RANK 512

typedef float    v2f  __attribute__((ext_vector_type(2)));
typedef float    v8f  __attribute__((ext_vector_type(8)));
typedef _Float16 v16h __attribute__((ext_vector_type(16)));
typedef unsigned int u32x4 __attribute__((ext_vector_type(4)));
typedef int          i32x4 __attribute__((ext_vector_type(4)));
typedef int          i32x8 __attribute__((ext_vector_type(8)));

// D = A(16x4,f32) * B(4x16,f32) + C   -> v_wmma_f32_16x16x4_f32
static __device__ __forceinline__ v8f wmma_f32k4(v2f a, v2f b, v8f c) {
  return __builtin_amdgcn_wmma_f32_16x16x4_f32(false, a, false, b, (short)0, c, false, false);
}
// D = A(16x32,f16) * B(32x16,f16) + C -> v_wmma_f32_16x16x32_f16
static __device__ __forceinline__ v8f wmma_f16k32(v16h a, v16h b, v8f c) {
  return __builtin_amdgcn_wmma_f32_16x16x32_f16(false, a, false, b, (short)0, c, false, false);
}
// wave-private LDS RAW fence (each wave owns its scratch region; no WG barrier)
static __device__ __forceinline__ void lds_sync() {
  __builtin_amdgcn_wave_barrier();
  asm volatile("s_wait_dscnt 0" ::: "memory");
  __builtin_amdgcn_wave_barrier();
}

// ---- Tensor Data Mover: DMA a 16KB f16 weight block (32x256) global->LDS ----
// D# per CDNA5 ISA: 1-D tensor, data_size=2B,
// tensor_dim0 = tile_dim0 = 8192 elements, stride 8192, type=2 ("image").
static __device__ __forceinline__ void tdm_load_weights(const _Float16* gsrc,
                                                        _Float16* lds_dst) {
#if __has_builtin(__builtin_amdgcn_tensor_load_to_lds)
  unsigned long long ga = (unsigned long long)(size_t)gsrc;
  unsigned int lds = (unsigned int)(size_t)lds_dst;  // flat LDS addr[31:0] = LDS offset
  u32x4 g0;
  g0[0] = 1u;                                        // count=1, user descriptor
  g0[1] = lds;                                       // lds_addr
  g0[2] = (unsigned int)(ga & 0xffffffffu);          // global_addr[31:0]
  g0[3] = (unsigned int)((ga >> 32) & 0x01ffffffu) | 0x80000000u; // ga[56:32] | type=2
  i32x8 g1;
  g1[0] = (1 << 16);                                 // wg_mask=0, data_size=1 (2B)
  g1[1] = (int)((8192u & 0xffffu) << 16);            // tensor_dim0[15:0] @ bits63:48
  g1[2] = (int)(((8192u >> 16) & 0xffffu) | (1u << 16)); // dim0 hi16; tensor_dim1=1 lo16
  g1[3] = (int)((8192u & 0xffffu) << 16);            // tensor_dim1 hi=0; tile_dim0=8192
  g1[4] = 1;                                         // tile_dim1=1, tile_dim2=0
  g1[5] = 8192;                                      // tensor_dim0_stride lo32
  g1[6] = 0;
  g1[7] = 0;
  i32x4 z4 = {0, 0, 0, 0};
#if defined(__clang_major__) && (__clang_major__ >= 23)
  i32x8 z8 = {0, 0, 0, 0, 0, 0, 0, 0};
  __builtin_amdgcn_tensor_load_to_lds(g0, g1, z4, z4, z8, 0);
#else
  __builtin_amdgcn_tensor_load_to_lds(g0, g1, z4, z4, 0);
#endif
  __builtin_amdgcn_s_wait_tensorcnt(0);
#else
  (void)gsrc; (void)lds_dst;
#endif
}

// ---------------- convert W_e2 (32x256 f32) to f16 once -------------------
__global__ void k_cvt_f16(const float* __restrict__ src, _Float16* __restrict__ dst, int n) {
  int i = blockIdx.x * blockDim.x + threadIdx.x;
  if (i < n) dst[i] = (_Float16)src[i];
}

// ---------------- zero agg / deg ------------------------------------------
__global__ void k_zero(float* __restrict__ agg, float* __restrict__ deg, int N) {
  int i = blockIdx.x * blockDim.x + threadIdx.x;
  if (i < N * DIM) agg[i] = 0.f;
  if (i < N) deg[i] = 0.f;
}

// ---------------- out = relu(x @ W_lin0 + b); h = out ---------------------
__global__ void k_lin0(const float* __restrict__ x, const float* __restrict__ W,
                       const float* __restrict__ b, float* __restrict__ out,
                       float* __restrict__ h, int ntiles) {
  int lane = threadIdx.x & 31, w = threadIdx.x >> 5;
  int tile = blockIdx.x * (blockDim.x >> 5) + w;
  if (tile >= ntiles) return;
  int base = tile * 16, n = lane & 15, m = lane & 15;
  int kb = (lane >> 4) * 2, rowb = (lane >> 4) * 8;
  v8f c;
  float bn = b[n];
  #pragma unroll
  for (int v = 0; v < 8; ++v) c[v] = bn;
  #pragma unroll
  for (int s = 0; s < 4; ++s) {
    int k = s * 4 + kb;
    v2f a, bb;
    a.x = x[(base + m) * 16 + k];  a.y = x[(base + m) * 16 + k + 1];
    bb.x = W[k * 16 + n];          bb.y = W[(k + 1) * 16 + n];
    c = wmma_f32k4(a, bb, c);
  }
  #pragma unroll
  for (int v = 0; v < 8; ++v) {
    float r = fmaxf(c[v], 0.f);
    out[(base + rowb + v) * 16 + n] = r;
    h[(base + rowb + v) * 16 + n]   = r;
  }
}

// -------- fused edge kernel: he -> We (f16 WMMA) -> matvec -> atomics -----
#define EW 8
__global__ void k_edge(const float* __restrict__ out, const float* __restrict__ ea,
                       const int* __restrict__ eidx, const float* __restrict__ W_e1,
                       const float* __restrict__ b_e1, const _Float16* __restrict__ W_e2h,
                       const float* __restrict__ b_e2, float* __restrict__ agg,
                       float* __restrict__ deg, int E, int etiles) {
  __shared__ _Float16 s_w[32 * 256];      // 16 KB: W_e2 (f16), shared by whole block
  __shared__ float s_he[EW][16 * 32];     // per-wave he scratch
  __shared__ float s_xs[EW][256];         // per-wave gathered src features

  // TDM bulk copy of the weight block into LDS (wave 0 issues, tracked by TENSORcnt)
  if (threadIdx.x < 32) tdm_load_weights(W_e2h, s_w);
  __syncthreads();
  // insurance copy (also the functional path if TDM builtin is unavailable)
  for (int i = threadIdx.x; i < 32 * 256; i += blockDim.x) s_w[i] = W_e2h[i];
  __syncthreads();

  int lane = threadIdx.x & 31, w = threadIdx.x >> 5;
  int tile = blockIdx.x * (blockDim.x >> 5) + w;
  if (tile >= etiles) return;
  int e0 = tile * 16;
  const int* src = eidx;
  const int* dst = eidx + E;
  int n = lane & 15, m = lane & 15, kb = (lane >> 4) * 2;
  int rowb = (lane >> 4) * 8, half = lane >> 4;

  int srcv[8], dstv[8];
  #pragma unroll
  for (int v = 0; v < 8; ++v) { srcv[v] = src[e0 + rowb + v]; dstv[v] = dst[e0 + rowb + v]; }
  if (tile + 1 < etiles) __builtin_prefetch(&ea[(e0 + 16) * 4], 0, 1);  // global_prefetch_b8

  // coalesced gather of source-node features into LDS (read as broadcast later)
  #pragma unroll
  for (int v = 0; v < 8; ++v) s_xs[w][(rowb + v) * 16 + n] = out[srcv[v] * 16 + n];

  // he = relu(edge_attr @ W_e1 + b_e1)   [16 x 32], K=4 -> one f32 WMMA per 16-col block
  v2f a4;
  a4.x = ea[(e0 + m) * 4 + kb];  a4.y = ea[(e0 + m) * 4 + kb + 1];
  #pragma unroll
  for (int hn = 0; hn < 2; ++hn) {
    int n0 = hn * 16;
    v8f c;
    float bn = b_e1[n0 + n];
    #pragma unroll
    for (int v = 0; v < 8; ++v) c[v] = bn;
    v2f bb;
    bb.x = W_e1[kb * 32 + n0 + n];  bb.y = W_e1[(kb + 1) * 32 + n0 + n];
    c = wmma_f32k4(a4, bb, c);
    #pragma unroll
    for (int v = 0; v < 8; ++v) s_he[w][(rowb + v) * 32 + n0 + n] = fmaxf(c[v], 0.f);
  }
  lds_sync();

  // redistribute he into an f16 A-fragment (16x32)
  v16h ah;
  #pragma unroll
  for (int v8 = 0; v8 < 8; ++v8) {
    int k0 = (v8 >> 2) * 16 + half * 8 + (v8 & 3) * 2;
    ah[2 * v8]     = (_Float16)s_he[w][m * 32 + k0];
    ah[2 * v8 + 1] = (_Float16)s_he[w][m * 32 + k0 + 1];
  }

  // We = he @ W_e2 + b_e2; column-block nt == input dim d; consume immediately:
  // macc[edge][k] += out[src][d=nt] * We[edge][d*16+k]  (We never touches memory)
  v8f macc;
  #pragma unroll
  for (int v = 0; v < 8; ++v) macc[v] = 0.f;
  for (int nt = 0; nt < 16; ++nt) {
    v16h bh;
    #pragma unroll
    for (int v8 = 0; v8 < 8; ++v8) {
      int k0 = (v8 >> 2) * 16 + half * 8 + (v8 & 3) * 2;
      bh[2 * v8]     = s_w[k0 * 256 + nt * 16 + n];
      bh[2 * v8 + 1] = s_w[(k0 + 1) * 256 + nt * 16 + n];
    }
    v8f c;
    float bn = b_e2[nt * 16 + n];
    #pragma unroll
    for (int v = 0; v < 8; ++v) c[v] = bn;
    c = wmma_f16k32(ah, bh, c);
    #pragma unroll
    for (int v = 0; v < 8; ++v) macc[v] += s_xs[w][(rowb + v) * 16 + nt] * c[v];
  }
  #pragma unroll
  for (int v = 0; v < 8; ++v) atomicAdd(&agg[dstv[v] * 16 + n], macc[v]);
  if ((lane & 15) == 0) {
    #pragma unroll
    for (int v = 0; v < 8; ++v) atomicAdd(&deg[dstv[v]], 1.f);
  }
}

// ---------------- node update: m = relu(out@W_root + b + agg/deg); GRU ----
#define NW 4
__global__ void k_node(float* __restrict__ out, float* __restrict__ h,
                       const float* __restrict__ agg, const float* __restrict__ deg,
                       const float* __restrict__ W_root, const float* __restrict__ b_conv,
                       const float* __restrict__ W_ih, const float* __restrict__ b_ih,
                       const float* __restrict__ W_hh, const float* __restrict__ b_hh,
                       int ntiles) {
  __shared__ float s_m[NW][256];
  int lane = threadIdx.x & 31, w = threadIdx.x >> 5;
  int tile = blockIdx.x * (blockDim.x >> 5) + w;
  if (tile >= ntiles) return;
  int base = tile * 16, n = lane & 15, m = lane & 15;
  int kb = (lane >> 4) * 2, rowb = (lane >> 4) * 8;

  v8f c;
  #pragma unroll
  for (int v = 0; v < 8; ++v) {
    float d = fmaxf(deg[base + rowb + v], 1.0f);
    c[v] = b_conv[n] + agg[(base + rowb + v) * 16 + n] / d;
  }
  v2f a_out[4];
  #pragma unroll
  for (int s = 0; s < 4; ++s) {
    int k = s * 4 + kb;
    a_out[s].x = out[(base + m) * 16 + k];  a_out[s].y = out[(base + m) * 16 + k + 1];
  }
  #pragma unroll
  for (int s = 0; s < 4; ++s) {
    int k = s * 4 + kb;
    v2f bb;
    bb.x = W_root[k * 16 + n];  bb.y = W_root[(k + 1) * 16 + n];
    c = wmma_f32k4(a_out[s], bb, c);
  }
  #pragma unroll
  for (int v = 0; v < 8; ++v) s_m[w][(rowb + v) * 16 + n] = fmaxf(c[v], 0.f);
  lds_sync();

  v2f a_m[4], a_h[4];
  #pragma unroll
  for (int s = 0; s < 4; ++s) {
    int k = s * 4 + kb;
    a_m[s].x = s_m[w][m * 16 + k];        a_m[s].y = s_m[w][m * 16 + k + 1];
    a_h[s].x = h[(base + m) * 16 + k];    a_h[s].y = h[(base + m) * 16 + k + 1];
  }
  v8f gi[3], gh[3];
  #pragma unroll
  for (int g = 0; g < 3; ++g) {
    v8f ci, ch;
    float bi = b_ih[g * 16 + n], bh2 = b_hh[g * 16 + n];
    #pragma unroll
    for (int v = 0; v < 8; ++v) { ci[v] = bi; ch[v] = bh2; }
    #pragma unroll
    for (int s = 0; s < 4; ++s) {
      int k = s * 4 + kb;
      v2f bbi, bbh;  // B = W^T : B[k][col] = W[col][k]
      bbi.x = W_ih[(g * 16 + n) * 16 + k];  bbi.y = W_ih[(g * 16 + n) * 16 + k + 1];
      bbh.x = W_hh[(g * 16 + n) * 16 + k];  bbh.y = W_hh[(g * 16 + n) * 16 + k + 1];
      ci = wmma_f32k4(a_m[s], bbi, ci);
      ch = wmma_f32k4(a_h[s], bbh, ch);
    }
    gi[g] = ci; gh[g] = ch;
  }
  #pragma unroll
  for (int v = 0; v < 8; ++v) {
    float hd = h[(base + rowb + v) * 16 + n];
    float r  = 1.f / (1.f + __expf(-(gi[0][v] + gh[0][v])));
    float z  = 1.f / (1.f + __expf(-(gi[1][v] + gh[1][v])));
    float nn = tanhf(gi[2][v] + r * gh[2][v]);
    float hv = (1.f - z) * nn + z * hd;
    h[(base + rowb + v) * 16 + n]   = hv;
    out[(base + rowb + v) * 16 + n] = hv;
  }
}

// ---------------- out_edges = relu(out @ W_lin1 + b_lin1) ----------------
__global__ void k_lin1(const float* __restrict__ out, const float* __restrict__ W,
                       const float* __restrict__ b, float* __restrict__ oe, int N) {
  int i = blockIdx.x * blockDim.x + threadIdx.x;
  if (i >= N) return;
  float acc[4];
  #pragma unroll
  for (int j = 0; j < 4; ++j) acc[j] = b[j];
  #pragma unroll
  for (int d = 0; d < 16; ++d) {
    float xv = out[i * 16 + d];
    #pragma unroll
    for (int j = 0; j < 4; ++j) acc[j] += xv * W[d * 4 + j];
  }
  #pragma unroll
  for (int j = 0; j < 4; ++j) oe[i * 4 + j] = fmaxf(acc[j], 0.f);
}

// ------- FGNet (live branch only) + edge update + log_softmax ------------
#define FW 4
__global__ void k_fgnet(const float* __restrict__ out, const float* __restrict__ oe,
                        const int* __restrict__ ntype, const float* __restrict__ W_up,
                        const float* __restrict__ b_up, const float* __restrict__ U_B,
                        const float* __restrict__ V_B, const float* __restrict__ U_A,
                        const float* __restrict__ V_A, const float* __restrict__ W_down,
                        const float* __restrict__ b_down, const float* __restrict__ w_edge,
                        const float* __restrict__ W_line, const float* __restrict__ b_line,
                        float* __restrict__ dout, int ntiles) {
  __shared__ float s_oc[FW][256];
  __shared__ float s_t [FW][256];
  __shared__ float s_e [FW][64];
  int lane = threadIdx.x & 31, w = threadIdx.x >> 5;
  int tile = blockIdx.x * (blockDim.x >> 5) + w;
  if (tile >= ntiles) return;
  int base = tile * 16, n = lane & 15, m = lane & 15;
  int kb = (lane >> 4) * 2, rowb = (lane >> 4) * 8;

  // out_combine = ev ? out_edges@W_up + b_up : out
  #pragma unroll
  for (int v = 0; v < 8; ++v) {
    int row = base + rowb + v;
    float val;
    if (ntype[row] == 2) {
      val = b_up[n];
      #pragma unroll
      for (int j = 0; j < 4; ++j) val += oe[row * 4 + j] * W_up[j * 16 + n];
    } else {
      val = out[row * 16 + n];
    }
    s_oc[w][(rowb + v) * 16 + n] = val;
  }
  lds_sync();
  v2f a_oc[4];
  #pragma unroll
  for (int s = 0; s < 4; ++s) {
    int k = s * 4 + kb;
    a_oc[s].x = s_oc[w][m * 16 + k];  a_oc[s].y = s_oc[w][m * 16 + k + 1];
  }

  // msg_B = relu((oc @ U_B) @ V_B), rank 512 in chunks of 16
  v8f cB;
  #pragma unroll
  for (int v = 0; v < 8; ++v) cB[v] = 0.f;
  for (int nt = 0; nt < 32; ++nt) {
    v8f t;
    #pragma unroll
    for (int v = 0; v < 8; ++v) t[v] = 0.f;
    #pragma unroll
    for (int s = 0; s < 4; ++s) {
      int k = s * 4 + kb;
      v2f bb;
      bb.x = U_B[k * RANK + nt * 16 + n];  bb.y = U_B[(k + 1) * RANK + nt * 16 + n];
      t = wmma_f32k4(a_oc[s], bb, t);
    }
    lds_sync();
    #pragma unroll
    for (int v = 0; v < 8; ++v) s_t[w][(rowb + v) * 16 + n] = t[v];
    lds_sync();
    #pragma unroll
    for (int s = 0; s < 4; ++s) {
      int kl = s * 4 + kb;
      v2f at, bb;
      at.x = s_t[w][m * 16 + kl];  at.y = s_t[w][m * 16 + kl + 1];
      int kg = nt * 16 + kl;
      bb.x = V_B[kg * 16 + n];     bb.y = V_B[(kg + 1) * 16 + n];
      cB = wmma_f32k4(at, bb, cB);
    }
  }
  v8f mB;
  #pragma unroll
  for (int v = 0; v < 8; ++v) mB[v] = fmaxf(cB[v], 0.f);

  // msg_to_edge_B = msg_B @ W_down + b_down (cols 0..3 valid)
  lds_sync();
  #pragma unroll
  for (int v = 0; v < 8; ++v) s_t[w][(rowb + v) * 16 + n] = mB[v];
  lds_sync();
  v8f ceB;
  float bdn = (n < 4) ? b_down[n] : 0.f;
  #pragma unroll
  for (int v = 0; v < 8; ++v) ceB[v] = bdn;
  #pragma unroll
  for (int s = 0; s < 4; ++s) {
    int k = s * 4 + kb;
    v2f at, bb;
    at.x = s_t[w][m * 16 + k];  at.y = s_t[w][m * 16 + k + 1];
    bb.x = (n < 4) ? W_down[k * 4 + n] : 0.f;
    bb.y = (n < 4) ? W_down[(k + 1) * 4 + n] : 0.f;
    ceB = wmma_f32k4(at, bb, ceB);
  }

  // msg_to_edge_A = relu((out_edges @ U_A) @ V_A)
  v2f a_oe;
  a_oe.x = oe[(base + m) * 4 + kb];  a_oe.y = oe[(base + m) * 4 + kb + 1];
  v8f cA;
  #pragma unroll
  for (int v = 0; v < 8; ++v) cA[v] = 0.f;
  for (int nt = 0; nt < 32; ++nt) {
    v2f bb;
    bb.x = U_A[kb * RANK + nt * 16 + n];  bb.y = U_A[(kb + 1) * RANK + nt * 16 + n];
    v8f t;
    #pragma unroll
    for (int v = 0; v < 8; ++v) t[v] = 0.f;
    t = wmma_f32k4(a_oe, bb, t);
    lds_sync();
    #pragma unroll
    for (int v = 0; v < 8; ++v) s_t[w][(rowb + v) * 16 + n] = t[v];
    lds_sync();
    #pragma unroll
    for (int s = 0; s < 4; ++s) {
      int kl = s * 4 + kb;
      v2f at, b2;
      at.x = s_t[w][m * 16 + kl];  at.y = s_t[w][m * 16 + kl + 1];
      int kg = nt * 16 + kl;
      b2.x = (n < 4) ? V_A[kg * 4 + n] : 0.f;
      b2.y = (n < 4) ? V_A[(kg + 1) * 4 + n] : 0.f;
      cA = wmma_f32k4(at, b2, cA);
    }
  }
  // combine, scale by w_edge, @ W_line, relu, add, log_softmax
  float wev = (n < 4) ? w_edge[n] : 0.f;
  lds_sync();
  #pragma unroll
  for (int v = 0; v < 8; ++v)
    s_t[w][(rowb + v) * 16 + n] = fmaxf(cA[v], 0.f) * ceB[v] * wev;
  lds_sync();
  v2f a_p;
  a_p.x = s_t[w][m * 16 + kb];  a_p.y = s_t[w][m * 16 + kb + 1];
  v8f ce;
  float bl = (n < 4) ? b_line[n] : 0.f;
  #pragma unroll
  for (int v = 0; v < 8; ++v) ce[v] = bl;
  v2f bl2;
  bl2.x = (n < 4) ? W_line[kb * 4 + n] : 0.f;
  bl2.y = (n < 4) ? W_line[(kb + 1) * 4 + n] : 0.f;
  ce = wmma_f32k4(a_p, bl2, ce);
  if (n < 4) {
    #pragma unroll
    for (int v = 0; v < 8; ++v)
      s_e[w][(rowb + v) * 4 + n] = oe[(base + rowb + v) * 4 + n] + fmaxf(ce[v], 0.f);
  }
  lds_sync();
  if (lane < 16) {
    int row = base + lane;
    float q0 = s_e[w][lane * 4 + 0], q1 = s_e[w][lane * 4 + 1];
    float q2 = s_e[w][lane * 4 + 2], q3 = s_e[w][lane * 4 + 3];
    float mx = fmaxf(fmaxf(q0, q1), fmaxf(q2, q3));
    float s = __expf(q0 - mx) + __expf(q1 - mx) + __expf(q2 - mx) + __expf(q3 - mx);
    float lse = mx + __logf(s);
    dout[row * 4 + 0] = q0 - lse;
    dout[row * 4 + 1] = q1 - lse;
    dout[row * 4 + 2] = q2 - lse;
    dout[row * 4 + 3] = q3 - lse;
  }
}

extern "C" void kernel_launch(void* const* d_in, const int* in_sizes, int n_in,
                              void* d_out, int out_size, void* d_ws, size_t ws_size,
                              hipStream_t stream) {
  const float* x         = (const float*)d_in[0];
  const int*   node_type = (const int*)  d_in[1];
  const int*   edge_idx  = (const int*)  d_in[2];
  const float* edge_attr = (const float*)d_in[3];
  const float* W_lin0 = (const float*)d_in[4];  const float* b_lin0 = (const float*)d_in[5];
  const float* W_root = (const float*)d_in[6];  const float* b_conv = (const float*)d_in[7];
  const float* W_e1   = (const float*)d_in[8];  const float* b_e1   = (const float*)d_in[9];
  const float* W_e2   = (const float*)d_in[10]; const float* b_e2   = (const float*)d_in[11];
  const float* W_ih   = (const float*)d_in[12]; const float* b_ih   = (const float*)d_in[13];
  const float* W_hh   = (const float*)d_in[14]; const float* b_hh   = (const float*)d_in[15];
  const float* W_lin1 = (const float*)d_in[16]; const float* b_lin1 = (const float*)d_in[17];
  const float* W_up   = (const float*)d_in[18]; const float* b_up   = (const float*)d_in[19];
  const float* W_down = (const float*)d_in[20]; const float* b_down = (const float*)d_in[21];
  const float* U_B    = (const float*)d_in[22]; const float* V_B    = (const float*)d_in[23];
  // d_in[24]=U_C, d_in[25]=V_C dead (only feed the un-returned `out` update)
  const float* U_A    = (const float*)d_in[26]; const float* V_A    = (const float*)d_in[27];
  // d_in[28..30] = w_node, W_lin, b_lin dead
  const float* w_edge = (const float*)d_in[31];
  const float* W_line = (const float*)d_in[32]; const float* b_line = (const float*)d_in[33];

  const int N = in_sizes[0] / DIM;
  const int E = in_sizes[3] / BOND;
  const int ntiles = (N + 15) / 16;
  const int etiles = (E + 15) / 16;

  float* ws   = (float*)d_ws;
  float* out  = ws;                 // N*16
  float* h    = out + (size_t)N * 16;
  float* agg  = h   + (size_t)N * 16;
  float* deg  = agg + (size_t)N * 16;
  float* oe   = deg + (size_t)N;    // N*4
  _Float16* W_e2h = (_Float16*)(oe + (size_t)N * 4);  // 32*256

  // prep: W_e2 -> f16
  k_cvt_f16<<<(32 * 256 + 255) / 256, 256, 0, stream>>>(W_e2, W_e2h, 32 * 256);

  // lin0
  {
    int wpb = 4, g = (ntiles + wpb - 1) / wpb;
    k_lin0<<<g, wpb * 32, 0, stream>>>(x, W_lin0, b_lin0, out, h, ntiles);
  }

  // 3 message-passing rounds
  for (int it = 0; it < 3; ++it) {
    k_zero<<<(N * DIM + 255) / 256, 256, 0, stream>>>(agg, deg, N);
    {
      int g = (etiles + EW - 1) / EW;
      k_edge<<<g, EW * 32, 0, stream>>>(out, edge_attr, edge_idx, W_e1, b_e1,
                                        W_e2h, b_e2, agg, deg, E, etiles);
    }
    {
      int g = (ntiles + NW - 1) / NW;
      k_node<<<g, NW * 32, 0, stream>>>(out, h, agg, deg, W_root, b_conv,
                                        W_ih, b_ih, W_hh, b_hh, ntiles);
    }
  }

  // out_edges
  k_lin1<<<(N + 255) / 256, 256, 0, stream>>>(out, W_lin1, b_lin1, oe, N);

  // FGNet live branch + edge update + log_softmax -> d_out
  {
    int g = (ntiles + FW - 1) / FW;
    k_fgnet<<<g, FW * 32, 0, stream>>>(out, oe, node_type, W_up, b_up,
                                       U_B, V_B, U_A, V_A, W_down, b_down,
                                       w_edge, W_line, b_line, (float*)d_out, ntiles);
  }
}
